// BSRNN_23433341567760
// MI455X (gfx1250) — compile-verified, matrix operand order checked by reading, not run on hardware
//
#include <hip/hip_runtime.h>
#include <hip/hip_bf16.h>

// ---------------------------------------------------------------------------
// BSRNN on MI455X (gfx1250): all GEMM-like work routed through
// v_wmma_f32_16x16x32_bf16 (fp32 accumulate). wave32 / WGP model.
// Vectorized (b64/b128) load paths; branch-free edge handling.
// ---------------------------------------------------------------------------

typedef __bf16 bf16_t;
typedef __attribute__((ext_vector_type(16))) __bf16          v16bf;
typedef __attribute__((ext_vector_type(8)))  float           v8f;
typedef __attribute__((ext_vector_type(4)))  float           v4f;
typedef __attribute__((ext_vector_type(2)))  float           v2f;
typedef __attribute__((ext_vector_type(8)))  unsigned short  v8us;
typedef __attribute__((ext_vector_type(4)))  unsigned int    v4u;
typedef __attribute__((ext_vector_type(2)))  unsigned int    v2u;

union FragU { v8us h[2]; v16bf v; };

#define WMMA_BF16(A, B, C) \
    __builtin_amdgcn_wmma_f32_16x16x32_bf16(false, (A), false, (B), (short)0, (C), false, false)

__device__ __forceinline__ unsigned short f2bf_us(float f) {
    union { float f; unsigned u; } x; x.f = f;
    unsigned r = x.u + 0x7FFFu + ((x.u >> 16) & 1u);   // round-to-nearest-even
    return (unsigned short)(r >> 16);
}
__device__ __forceinline__ unsigned pack2(float a, float b) {
    return (unsigned)f2bf_us(a) | ((unsigned)f2bf_us(b) << 16);
}
__device__ __forceinline__ bf16_t us2bf(unsigned short s) {
    union { unsigned short u; bf16_t b; } y; y.u = s; return y.b;
}
__device__ __forceinline__ bf16_t f2bf(float f) { return us2bf(f2bf_us(f)); }
__device__ __forceinline__ float sigm(float x) { return 1.f / (1.f + __expf(-x)); }

#define C_   4
#define F_   2050
#define T_   2000
#define NB_  12
#define D_   64
#define HR_  768        // NB_ * D_

// ---------------------------------------------------------------------------
// Generic GEMM:  Cout[M,N] = act( A[M,K] * W[N,K]^T + bias (+bias2) (+Res) )
// 64x64 block, 8 waves (4 x 2), BK = 64 (two bf16 WMMA K steps / iter).
// Fast path: v2f global loads -> bf16 pack -> b128 LDS stores (all strides
// in this model are even, so 8B alignment is guaranteed).
// ---------------------------------------------------------------------------
#define GBM 64
#define GBN 64
#define GBK 64
#define GLS 72   // LDS row stride in halves: 144 B (16B-aligned rows, padded)

__global__ __launch_bounds__(256)
void k_gemm(const float* __restrict__ A, int lda,
            const float* __restrict__ W,                 // (N,K) row-major
            const float* __restrict__ bias,
            const float* __restrict__ bias2,
            const float* __restrict__ Res, int ldr,
            float* __restrict__ Cout, int ldc,
            int M, int N, int K, int act)
{
    __shared__ unsigned short As[GBM * GLS];
    __shared__ unsigned short Bs[GBN * GLS];

    const int bm  = blockIdx.y * GBM;
    const int bn  = blockIdx.x * GBN;
    const int tid = threadIdx.x;
    const int wave = tid >> 5, lane = tid & 31;
    const int wm = wave >> 1, wn = wave & 1;     // 4x2 wave grid
    const int l16 = lane & 15, hi = lane >> 4;
    const bool fullMN = (bm + GBM <= M) && (bn + GBN <= N) && (((lda | K) & 1) == 0);

    v8f acc0 = {0,0,0,0,0,0,0,0};
    v8f acc1 = {0,0,0,0,0,0,0,0};

    const int nKb = (K + GBK - 1) / GBK;
    for (int kb = 0; kb < nKb; ++kb) {
        const int k0 = kb * GBK;

        if (fullMN && (k0 + GBK <= K)) {
            // -------- vectorized tile stage: 16 floats / thread / matrix ----
            const int r  = tid >> 2;          // 0..63
            const int kk = (tid & 3) * 16;    // 0,16,32,48
            const float* pa = A + (long)(bm + r) * lda + k0 + kk;
            const float* pw = W + (long)(bn + r) * K  + k0 + kk;
            if (kb + 1 < nKb) {               // gfx1250 global_prefetch_b8
                __builtin_prefetch(pa + GBK, 0, 0);
                __builtin_prefetch(pw + GBK, 0, 0);
            }
            v2f fa[8], fw[8];
            #pragma unroll
            for (int j = 0; j < 8; ++j) {
                fa[j] = ((const v2f*)pa)[j];
                fw[j] = ((const v2f*)pw)[j];
            }
            v4u pa0, pa1, pw0, pw1;
            #pragma unroll
            for (int j = 0; j < 4; ++j) {
                pa0[j] = pack2(fa[j][0],     fa[j][1]);
                pa1[j] = pack2(fa[j + 4][0], fa[j + 4][1]);
                pw0[j] = pack2(fw[j][0],     fw[j][1]);
                pw1[j] = pack2(fw[j + 4][0], fw[j + 4][1]);
            }
            *reinterpret_cast<v4u*>(&As[r * GLS + kk])     = pa0;
            *reinterpret_cast<v4u*>(&As[r * GLS + kk + 8]) = pa1;
            *reinterpret_cast<v4u*>(&Bs[r * GLS + kk])     = pw0;
            *reinterpret_cast<v4u*>(&Bs[r * GLS + kk + 8]) = pw1;
        } else {
            // -------- edge/tail: clamped indices + select-to-zero (no branch)
            #pragma unroll
            for (int i = 0; i < 16; ++i) {
                int idx = (tid << 4) + i;
                int r  = idx >> 6;
                int kk = idx & 63;
                int gk = k0 + kk;
                int ra = bm + r; int ca = ra < M ? ra : M - 1;
                int rb = bn + r; int cb = rb < N ? rb : N - 1;
                int gc = gk < K ? gk : K - 1;
                float va = A[(long)ca * lda + gc];
                float vb = W[(long)cb * K  + gc];
                if (!(ra < M && gk < K)) va = 0.f;
                if (!(rb < N && gk < K)) vb = 0.f;
                As[r * GLS + kk] = f2bf_us(va);
                Bs[r * GLS + kk] = f2bf_us(vb);
            }
        }
        __syncthreads();

        // fragments: two contiguous 16B chunks each (ISA 7.12.2 layout)
        FragU a0, a1, b00, b01, b10, b11;
        const unsigned short* ap  = &As[(wm * 16 + l16)      * GLS + (hi ? 8 : 0)];
        const unsigned short* bp0 = &Bs[(wn * 32 + l16)      * GLS + (hi ? 8 : 0)];
        const unsigned short* bp1 = &Bs[(wn * 32 + 16 + l16) * GLS + (hi ? 8 : 0)];
        a0.h[0]  = *(const v8us*)(ap);       a0.h[1]  = *(const v8us*)(ap  + 16);
        a1.h[0]  = *(const v8us*)(ap  + 32); a1.h[1]  = *(const v8us*)(ap  + 48);
        b00.h[0] = *(const v8us*)(bp0);      b00.h[1] = *(const v8us*)(bp0 + 16);
        b01.h[0] = *(const v8us*)(bp0 + 32); b01.h[1] = *(const v8us*)(bp0 + 48);
        b10.h[0] = *(const v8us*)(bp1);      b10.h[1] = *(const v8us*)(bp1 + 16);
        b11.h[0] = *(const v8us*)(bp1 + 32); b11.h[1] = *(const v8us*)(bp1 + 48);

        acc0 = WMMA_BF16(a0.v, b00.v, acc0);
        acc0 = WMMA_BF16(a1.v, b01.v, acc0);
        acc1 = WMMA_BF16(a0.v, b10.v, acc1);
        acc1 = WMMA_BF16(a1.v, b11.v, acc1);
        __syncthreads();
    }

    // epilogue: C/D layout -> lane l16 = column, vgpr r + 8*hi = row
    v8f accs[2] = {acc0, acc1};
    #pragma unroll
    for (int t = 0; t < 2; ++t) {
        int col = bn + wn * 32 + t * 16 + l16;
        if (col >= N) continue;
        float bv = bias ? bias[col] : 0.f;
        if (bias2) bv += bias2[col];
        #pragma unroll
        for (int r = 0; r < 8; ++r) {
            int row = bm + wm * 16 + hi * 8 + r;
            if (row >= M) continue;
            float v = accs[t][r] + bv;
            if (Res) v += Res[(long)row * ldr + col];
            if (act) v = (v > 0.f) ? v : 0.01f * v;     // leaky_relu(0.01)
            Cout[(long)row * ldc + col] = v;
        }
    }
}

// ---------------------------------------------------------------------------
// Fused LSTM scan. One WG = 16 sequences. Per step:
//   gates = XG[:,t,:] + h @ Whh^T   (16x64 @ 64x256, 4 WMMAs per wave)
// Whh pre-loaded once as register-resident WMMA B-fragments; h as bf16 in
// LDS (b128 fragment loads); c persistent per thread; float4 gate I/O.
// ---------------------------------------------------------------------------
#define HLS 72   // Hs row stride in halves (144 B, 16B-aligned)

__global__ __launch_bounds__(256)
void k_lstm_scan(const float* __restrict__ XG,    // (B, L, 256)
                 const float* __restrict__ Whh,   // (256, 64) row-major
                 float* __restrict__ Hout, int ldo, int colOff,
                 int B, int L, int reverse)
{
    __shared__ unsigned short Hs[16 * HLS];  // h (bf16)
    __shared__ float Gs[16 * 256];           // recurrent gate partials

    const int tid = threadIdx.x;
    const int wave = tid >> 5, lane = tid & 31;
    const int l16 = lane & 15, hi = lane >> 4;
    const int rowBase = blockIdx.x * 16;

    // Whh^T fragments: wave owns gate columns [wave*32, wave*32+32)
    v16bf b00, b01, b10, b11;
    {
        const int n0 = wave * 32 + l16;
        const int n1 = n0 + 16;
        #pragma unroll
        for (int e = 0; e < 16; ++e) {
            int k0 = (hi ? 8 : 0) + e + (e >= 8 ? 8 : 0);  // K-tile 0
            int k1 = 32 + k0;                              // K-tile 1
            b00[e] = f2bf(Whh[n0 * 64 + k0]);
            b01[e] = f2bf(Whh[n0 * 64 + k1]);
            b10[e] = f2bf(Whh[n1 * 64 + k0]);
            b11[e] = f2bf(Whh[n1 * 64 + k1]);
        }
    }

    for (int i = tid; i < 16 * HLS; i += 256) Hs[i] = 0;   // h0 = 0
    float c0 = 0.f, c1 = 0.f, c2 = 0.f, c3 = 0.f;          // c0 = 0
    const int r_el = tid >> 4;          // row 0..15 for elementwise phase
    const int u_el = (tid & 15) * 4;    // hidden base 0..60
    __syncthreads();

    for (int s = 0; s < L; ++s) {
        const int t = reverse ? (L - 1 - s) : s;

        // A fragments from current h: b128 LDS loads
        FragU a0, a1;
        const unsigned short* hp = &Hs[l16 * HLS + (hi ? 8 : 0)];
        a0.h[0] = *(const v8us*)(hp);      a0.h[1] = *(const v8us*)(hp + 16);
        a1.h[0] = *(const v8us*)(hp + 32); a1.h[1] = *(const v8us*)(hp + 48);

        v8f acc0 = {0,0,0,0,0,0,0,0};
        v8f acc1 = {0,0,0,0,0,0,0,0};
        acc0 = WMMA_BF16(a0.v, b00, acc0);
        acc0 = WMMA_BF16(a1.v, b01, acc0);
        acc1 = WMMA_BF16(a0.v, b10, acc1);
        acc1 = WMMA_BF16(a1.v, b11, acc1);

        const int n0 = wave * 32 + l16, n1 = n0 + 16;
        #pragma unroll
        for (int r = 0; r < 8; ++r) {
            Gs[(hi * 8 + r) * 256 + n0] = acc0[r];
            Gs[(hi * 8 + r) * 256 + n1] = acc1[r];
        }
        __syncthreads();

        // gate nonlinearities + state update (i, f, g, o), float4 I/O
        const long xgb = ((long)(rowBase + r_el) * L + t) * 256;
        v4f gi = *(const v4f*)&Gs[r_el * 256 +       u_el];
        v4f gf = *(const v4f*)&Gs[r_el * 256 +  64 + u_el];
        v4f gg = *(const v4f*)&Gs[r_el * 256 + 128 + u_el];
        v4f go = *(const v4f*)&Gs[r_el * 256 + 192 + u_el];
        v4f xi = *(const v4f*)&XG[xgb +       u_el];
        v4f xf = *(const v4f*)&XG[xgb +  64 + u_el];
        v4f xg = *(const v4f*)&XG[xgb + 128 + u_el];
        v4f xo = *(const v4f*)&XG[xgb + 192 + u_el];

        v4f hv;
        #pragma unroll
        for (int q = 0; q < 4; ++q) {
            float ig = sigm (gi[q] + xi[q]);
            float fg = sigm (gf[q] + xf[q]);
            float g2 = tanhf(gg[q] + xg[q]);
            float og = sigm (go[q] + xo[q]);
            float& cc = (q == 0 ? c0 : q == 1 ? c1 : q == 2 ? c2 : c3);
            cc = fg * cc + ig * g2;
            hv[q] = og * tanhf(cc);
        }
        v2u hp2;
        hp2[0] = pack2(hv[0], hv[1]);
        hp2[1] = pack2(hv[2], hv[3]);
        *reinterpret_cast<v2u*>(&Hs[r_el * HLS + u_el]) = hp2;
        *(v4f*)&Hout[((long)(rowBase + r_el) * L + t) * ldo + colOff + u_el] = hv;
        __syncthreads();
    }
}

// ---------------------------------------------------------------------------
// Data-movement helpers
// ---------------------------------------------------------------------------
__global__ void k_band_tr(const float* __restrict__ x, float* __restrict__ out,
                          int f0, int d)               // (C,F,T) slice -> (C*T, d)
{
    long i = (long)blockIdx.x * 256 + threadIdx.x;
    long tot = (long)C_ * T_ * d;
    if (i >= tot) return;
    int k = (int)(i % d);
    long m = i / d;
    int t = (int)(m % T_);
    int c = (int)(m / T_);
    out[i] = x[((long)c * F_ + f0 + k) * T_ + t];
}

__global__ void k_zero_band(float* __restrict__ H, const float* __restrict__ zc)
{
    long i = (long)blockIdx.x * 256 + threadIdx.x;
    if (i >= (long)C_ * T_ * D_) return;
    int u = (int)(i & 63);
    long m = i >> 6;
    H[m * HR_ + 11 * D_ + u] = zc[u];
}

__global__ void k_perm_fwd(const float* __restrict__ in, float* __restrict__ out)
{   // (C,T,NB,D) -> (C,NB,T,D), float4 granularity
    long i = (long)blockIdx.x * 256 + threadIdx.x;
    if (i >= (long)C_ * T_ * NB_ * (D_ / 4)) return;
    int u = (int)(i & 15);
    long r = i >> 4;
    int n = (int)(r % NB_); r /= NB_;
    int t = (int)(r % T_);
    int c = (int)(r / T_);
    ((v4f*)out)[((((long)c * NB_ + n) * T_ + t) * (D_ / 4)) + u] = ((const v4f*)in)[i];
}

__global__ void k_perm_bwd(const float* __restrict__ in, float* __restrict__ out)
{   // (C,NB,T,D) -> (C,T,NB,D), float4 granularity
    long i = (long)blockIdx.x * 256 + threadIdx.x;
    if (i >= (long)C_ * T_ * NB_ * (D_ / 4)) return;
    int u = (int)(i & 15);
    long r = i >> 4;
    int t = (int)(r % T_); r /= T_;
    int n = (int)(r % NB_);
    int c = (int)(r / NB_);
    ((v4f*)out)[((((long)c * T_ + t) * NB_ + n) * (D_ / 4)) + u] = ((const v4f*)in)[i];
}

__global__ void k_apply_mask(const float* __restrict__ x,
                             const float* __restrict__ mask,  // (C*T, F)
                             float* __restrict__ out)
{
    long i = (long)blockIdx.x * 256 + threadIdx.x;
    if (i >= (long)C_ * F_ * T_) return;
    int t = (int)(i % T_);
    long r = i / T_;
    int f = (int)(r % F_);
    int c = (int)(r / F_);
    out[i] = x[i] * mask[((long)c * T_ + t) * F_ + f];
}

// ---------------------------------------------------------------------------
// Host orchestration
// ---------------------------------------------------------------------------
static inline int imax_h(int a, int b) { return a > b ? a : b; }

static void gemm_l(hipStream_t st, const float* A, int lda, const float* W,
                   const float* bias, const float* bias2,
                   const float* Res, int ldr,
                   float* Cc, int ldc, int M, int N, int K, int act)
{
    dim3 g((N + GBN - 1) / GBN, (M + GBM - 1) / GBM);
    k_gemm<<<g, 256, 0, st>>>(A, lda, W, bias, bias2, Res, ldr, Cc, ldc, M, N, K, act);
}

struct NrrP {
    const float *fcW, *fcb, *fiW, *fib;
    const float *Whh[2][2], *Wih[2][2], *bih[2][2], *bhh[2][2];
    int dirs;
};

static NrrP nrr_load(void* const* din, int base, int bidir)
{
    NrrP p; p.dirs = bidir ? 2 : 1;
    auto f = [&](int i) { return (const float*)din[i]; };
    // sorted-key pytree order: fc{W,b}, fc_in{W,b}, layers[l][d]{Whh,Wih,bhh,bih}
    p.fcW = f(base + 0); p.fcb = f(base + 1);
    p.fiW = f(base + 2); p.fib = f(base + 3);
    int k = base + 4;
    for (int l = 0; l < 2; ++l)
        for (int d = 0; d < p.dirs; ++d) {
            p.Whh[l][d] = f(k++);
            p.Wih[l][d] = f(k++);
            p.bhh[l][d] = f(k++);
            p.bih[l][d] = f(k++);
        }
    return p;
}

static void run_nrr(hipStream_t st, const float* X, float* OUT, int B, int L,
                    const NrrP& p, float* L0, float* L1, float* XG)
{
    const int M = B * L;
    gemm_l(st, X, 64, p.fiW, p.fib, nullptr, nullptr, 0, L0, 64, M, 64, 64, 0);
    const float* cur = L0; int curD = 64;
    for (int l = 0; l < 2; ++l) {
        float* ob = (l == 0) ? L1 : L0;
        for (int d = 0; d < p.dirs; ++d) {
            gemm_l(st, cur, curD, p.Wih[l][d], p.bih[l][d], p.bhh[l][d],
                   nullptr, 0, XG, 256, M, 256, curD, 0);
            k_lstm_scan<<<B / 16, 256, 0, st>>>(XG, p.Whh[l][d], ob,
                                                p.dirs * 64, d * 64, B, L, d == 1);
        }
        cur = ob; curD = p.dirs * 64;
    }
    gemm_l(st, cur, curD, p.fcW, p.fcb, nullptr, X, 64, OUT, 64, M, 64, curD, 0);
}

extern "C" void kernel_launch(void* const* d_in, const int* in_sizes, int n_in,
                              void* d_out, int out_size, void* d_ws, size_t ws_size,
                              hipStream_t stream)
{
    if (n_in < 286) return;   // x + 285 param leaves expected

    static const int WID[11] = {1, 2, 3, 6, 12, 24, 48, 96, 192, 384, 257};
    int DD[11], F0[11];
    { int acc = 0; for (int i = 0; i < 11; ++i) { DD[i] = 2 * WID[i]; F0[i] = acc; acc += DD[i]; } }

    const float* x = (const float*)d_in[0];
    auto Wp = [&](int i) { return (const float*)d_in[i]; };

    // Workspace layout (floats)
    float* ws   = (float*)d_ws;
    float* res  = ws;                         // 8000 * 2050 = 16,400,000
    float* H    = res  + 16400000L;           //  6,144,000 each
    float* HB   = H    +  6144000L;
    float* HS   = HB   +  6144000L;
    float* HT   = HS   +  6144000L;
    float* scrA = HT   +  6144000L;           // 24,576,000 : XG | buf0+buf1
    float* scrB = scrA + 24576000L;           // 24,576,000 : L0+L1 | mask
    float* buf0 = scrA;
    float* buf1 = scrA + 6144000L;
    float* XG   = scrA;
    float* L0b  = scrB;
    float* L1b  = scrB + 12288000L;
    float* mask = scrB;

    const int M8 = C_ * T_;   // 8000
    (void)in_sizes; (void)out_size; (void)ws_size;

    // ---------------- Stage A: band encoders -> H (8000, 12, 64) -----------
    for (int i = 0; i < 11; ++i) {
        const int bi = 1 + 20 * i;           // band leaves: back*3, fc*3, post*2, pre*2
        const int d = DD[i], f0 = F0[i], h1 = imax_h(d, 64);
        float* R = res + (long)M8 * f0;

        long tot = (long)M8 * d;
        k_band_tr<<<(int)((tot + 255) / 256), 256, 0, stream>>>(x, buf0, f0, d);
        gemm_l(stream, buf0, d, Wp(bi + 16), Wp(bi + 17), nullptr, nullptr, 0, buf1, d,  M8, d,  d, 1); // pre0
        gemm_l(stream, buf1, d, Wp(bi + 18), Wp(bi + 19), nullptr, nullptr, 0, R,    d,  M8, d,  d, 1); // pre1 -> residual
        gemm_l(stream, R,    d, Wp(bi +  6), Wp(bi +  7), nullptr, nullptr, 0, buf0, h1, M8, h1, d, 1); // fc0
        gemm_l(stream, buf0, h1,Wp(bi +  8), Wp(bi +  9), nullptr, nullptr, 0, buf1, 64, M8, 64, h1,1); // fc1
        gemm_l(stream, buf1, 64,Wp(bi + 10), Wp(bi + 11), nullptr, nullptr, 0, H + i * 64, HR_, M8, 64, 64, 0); // fc2
    }
    k_zero_band<<<(M8 * 64 + 255) / 256, 256, 0, stream>>>(H, Wp(285));

    // ---------------- Stage B: 4 NRR blocks --------------------------------
    NrrP n0 = nrr_load((void* const*)d_in, 221, 1);
    NrrP n1 = nrr_load((void* const*)d_in, 241, 0);
    NrrP n2 = nrr_load((void* const*)d_in, 253, 1);
    NrrP n3 = nrr_load((void* const*)d_in, 273, 0);

    const long HN4 = (long)C_ * T_ * NB_ * (D_ / 4);
    const int  PB  = (int)((HN4 + 255) / 256);

    run_nrr(stream, H, HB, M8, NB_, n0, L0b, L1b, XG);               // bandwise
    k_perm_fwd<<<PB, 256, 0, stream>>>(HB, HS);
    run_nrr(stream, HS, HT, C_ * NB_, T_, n1, L0b, L1b, XG);         // timewise
    k_perm_bwd<<<PB, 256, 0, stream>>>(HT, H);
    run_nrr(stream, H, HB, M8, NB_, n2, L0b, L1b, XG);               // bandwise
    k_perm_fwd<<<PB, 256, 0, stream>>>(HB, HS);
    run_nrr(stream, HS, HT, C_ * NB_, T_, n3, L0b, L1b, XG);         // timewise
    k_perm_bwd<<<PB, 256, 0, stream>>>(HT, H);

    // ---------------- Stage C: band decoders -> mask ------------------------
    for (int i = 0; i < 11; ++i) {
        const int bi = 1 + 20 * i;
        const int d = DD[i], f0 = F0[i], h2 = imax_h(d, 128);
        const float* R = res + (long)M8 * f0;

        gemm_l(stream, H + i * 64, HR_, Wp(bi + 0), Wp(bi + 1), nullptr, nullptr, 0, buf0, 128, M8, 128, 64, 1); // back0
        gemm_l(stream, buf0, 128, Wp(bi + 2), Wp(bi + 3), nullptr, nullptr, 0, buf1, h2, M8, h2, 128, 1);        // back1
        gemm_l(stream, buf1, h2,  Wp(bi + 4), Wp(bi + 5), nullptr, nullptr, 0, buf0, d,  M8, d,  h2, 1);         // back2
        gemm_l(stream, buf0, d,   Wp(bi +12), Wp(bi +13), nullptr, nullptr, 0, buf1, d,  M8, d,  d,  1);         // post0
        gemm_l(stream, buf1, d,   Wp(bi +14), Wp(bi +15), nullptr, R, d, mask + f0, F_, M8, d, d, 0);            // post1 + residual
    }

    // ---------------- Final: out = x * mask ---------------------------------
    long tot = (long)C_ * F_ * T_;
    k_apply_mask<<<(int)((tot + 255) / 256), 256, 0, stream>>>(x, mask, (float*)d_out);
}
